// ContentBasedAttention_32727650795942
// MI455X (gfx1250) — matrix operand order repaired and checked by reading, not compiled, not run on hardware
//
#include <hip/hip_runtime.h>

typedef __attribute__((ext_vector_type(16))) __bf16        v16bf;
typedef __attribute__((ext_vector_type(8)))  float         v8f;
typedef __attribute__((ext_vector_type(4)))  unsigned int  v4u;

#define B_      32
#define TV      4096
#define H_      1024
#define A_      512
#define TTILE   128
#define NCHUNK  (TV / TTILE)     // 32 Tv-chunks per batch
#define PSTRIDE 1040             // per-partial floats: c[1024], m, l (+pad)
#define KC_N    (H_ / 32)        // 32 k-chunks of 32

static __device__ __forceinline__ unsigned short f2bf(float f) {
  // fp32 -> bf16 round-to-nearest-even (fallback path)
  unsigned int u = __float_as_uint(f);
  u += 0x7FFFu + ((u >> 16) & 1u);
  return (unsigned short)(u >> 16);
}

static __device__ __forceinline__ unsigned int pkbf(float a, float b) {
#if __has_builtin(__builtin_amdgcn_cvt_pk_bf16_f32)
  auto r = __builtin_amdgcn_cvt_pk_bf16_f32(a, b);   // v_cvt_pk_bf16_f32
  unsigned int u;
  __builtin_memcpy(&u, &r, 4);
  return u;
#else
  return (unsigned int)f2bf(a) | ((unsigned int)f2bf(b) << 16);
#endif
}

static __device__ __forceinline__ float fast_tanh(float x) {
#if __has_builtin(__builtin_amdgcn_tanhf)
  return __builtin_amdgcn_tanhf(x);                  // v_tanh_f32
#elif __has_builtin(__builtin_amdgcn_tanh_f32)
  return __builtin_amdgcn_tanh_f32(x);
#else
  return tanhf(x);
#endif
}

// ---------------------------------------------------------------------------
// Kernel 0: qb[b][a] = bias[a] + sum_h query[b,h] * W_Q[h,a]   (tiny GEMM)
// ---------------------------------------------------------------------------
__global__ void __launch_bounds__(256)
qproj_kernel(const float* __restrict__ query, const float* __restrict__ W_Q,
             const float* __restrict__ bias, float* __restrict__ qb) {
  const int a = blockIdx.x * 256 + threadIdx.x;   // 0..511
  const int b = blockIdx.y;
  const float* q = query + b * H_;
  float s = bias[a];
  for (int h = 0; h < H_; ++h) s = fmaf(q[h], W_Q[h * A_ + a], s);
  qb[b * A_ + a] = s;
}

// ---------------------------------------------------------------------------
// Kernel 1: pack W_V (fp32 [H,A] row-major) into bf16 WMMA B-fragments.
// Fragment (kc, ct): per-lane 16 bf16: col = ct*16 + (lane&15),
//                    K = kc*32 + (lane>>4)*8 + (j&7) + (j>>3)*16   (j = 0..15)
// ---------------------------------------------------------------------------
__global__ void __launch_bounds__(256)
packWV_kernel(const float* __restrict__ W_V, unsigned short* __restrict__ frags) {
  const int idx  = blockIdx.x * 256 + threadIdx.x;   // 0 .. 524287
  const int j    = idx & 15;
  const int lane = (idx >> 4) & 31;
  const int ct   = (idx >> 9) & 31;
  const int kc   = idx >> 14;
  const int K    = kc * 32 + ((lane >> 4) * 8) + (j & 7) + ((j >> 3) * 16);
  const int col  = ct * 16 + (lane & 15);
  frags[idx] = f2bf(W_V[K * A_ + col]);
}

// ---------------------------------------------------------------------------
// Kernel 2: per (batch, Tv-chunk of 128 rows):
//   stage value tile as bf16 A-fragments in LDS (convert ONCE),
//   WMMA GEMM (value tile @ W_V) -> tanh(+qb) -> . fc_w -> local softmax ->
//   exp-weighted partial context. Writes (c[1024], m, l) per chunk.
// 256 threads = 8 waves; wave w owns rows [chunk*128 + 16w, +16).
// Dynamic LDS: 8 waves * 32 kc * 32 lanes * 32B = 256 KB (of 320 KB/WGP).
// ---------------------------------------------------------------------------
__global__ void __launch_bounds__(256)
attn_partial_kernel(const float* __restrict__ value,
                    const unsigned short* __restrict__ frags,
                    const float* __restrict__ qb,
                    const float* __restrict__ fcw,
                    float* __restrict__ part) {
  const int tid   = threadIdx.x;
  const int lane  = tid & 31;
  const int wave  = tid >> 5;
  const int lo    = lane & 15;
  const int hi    = lane >> 4;
  const int chunk = blockIdx.x;           // 0..31
  const int b     = blockIdx.y;           // 0..31
  const int rowbase = chunk * TTILE;
  const float* vb = value + (size_t)b * TV * H_;

  extern __shared__ unsigned int afrag[];     // [wave][kc][lane][8 dwords]
  __shared__ float red[8][16][16];
  __shared__ float sc[TTILE];
  __shared__ float wexp[TTILE];
  __shared__ float sm[2];

  // per-thread fragment slot: contiguous 1KB per (wave,kc) -> conflict-free
  v4u* myfrag = reinterpret_cast<v4u*>(afrag) + (size_t)(wave * KC_N * 32 + lane) * 2;

  // ---- stage: convert this thread's A-fragment data fp32 -> bf16 into LDS
  // lane hi-half selects the +8 K offset per the ISA 16-bit A 16x32 layout.
  const float* arow = vb + (size_t)(rowbase + wave * 16 + lo) * H_ + hi * 8;
#pragma unroll 4
  for (int kc = 0; kc < KC_N; ++kc) {
    const float4* ap4 = reinterpret_cast<const float4*>(arow + kc * 32);
    float4 x0 = ap4[0], x1 = ap4[1];      // K = kb .. kb+7
    float4 x2 = ap4[4], x3 = ap4[5];      // K = kb+16 .. kb+23
    v4u w0, w1;
    w0.x = pkbf(x0.x, x0.y); w0.y = pkbf(x0.z, x0.w);
    w0.z = pkbf(x1.x, x1.y); w0.w = pkbf(x1.z, x1.w);
    w1.x = pkbf(x2.x, x2.y); w1.y = pkbf(x2.z, x2.w);
    w1.z = pkbf(x3.x, x3.y); w1.w = pkbf(x3.z, x3.w);
    myfrag[kc * 64 + 0] = w0;             // 2x ds_store_b128
    myfrag[kc * 64 + 1] = w1;
  }
  // Each thread reads back only its own LDS slots; same-wave LDS ops are
  // in-order, so no block barrier is required before the GEMM.

  float spart[8] = {0.f, 0.f, 0.f, 0.f, 0.f, 0.f, 0.f, 0.f};

  for (int a0 = 0; a0 < A_; a0 += 64) {
    v8f acc[4];
#pragma unroll
    for (int j = 0; j < 4; ++j)
#pragma unroll
      for (int r = 0; r < 8; ++r) acc[j][r] = 0.f;

#pragma unroll 4
    for (int kc = 0; kc < KC_N; ++kc) {
      // A fragment from LDS as two b128 loads
      union { v4u u[2]; v16bf v; } af;
      af.u[0] = myfrag[kc * 64 + 0];
      af.u[1] = myfrag[kc * 64 + 1];

      // B fragments: pre-packed bf16, one 32B load each (2x b128), L0/L2-hot
      const v16bf* bp = reinterpret_cast<const v16bf*>(frags) +
                        ((size_t)(kc * 32 + (a0 >> 4)) * 32 + lane);
      v16bf b0 = bp[0 * 32], b1 = bp[1 * 32], b2 = bp[2 * 32], b3 = bp[3 * 32];

      acc[0] = __builtin_amdgcn_wmma_f32_16x16x32_bf16(false, af.v, false, b0,
                                                       (short)0, acc[0], false, false);
      acc[1] = __builtin_amdgcn_wmma_f32_16x16x32_bf16(false, af.v, false, b1,
                                                       (short)0, acc[1], false, false);
      acc[2] = __builtin_amdgcn_wmma_f32_16x16x32_bf16(false, af.v, false, b2,
                                                       (short)0, acc[2], false, false);
      acc[3] = __builtin_amdgcn_wmma_f32_16x16x32_bf16(false, af.v, false, b3,
                                                       (short)0, acc[3], false, false);
    }

    // --- epilogue: energy = tanh(acc + qb), score += energy * fc_w
#pragma unroll
    for (int j = 0; j < 4; ++j) {
      const int col = a0 + 16 * j + lo;     // C/D: N = lane&15
      const float qv = qb[b * A_ + col];
      const float fw = fcw[col];
#pragma unroll
      for (int r = 0; r < 8; ++r) {         // C/D: M = r + 8*hi
        float e = fast_tanh(acc[j][r] + qv);
        spart[r] = fmaf(e, fw, spart[r]);
      }
    }
  }

  // --- reduce scores across the 16 N-lanes per row
#pragma unroll
  for (int r = 0; r < 8; ++r) red[wave][r + 8 * hi][lo] = spart[r];
  __syncthreads();
  if (tid < TTILE) {
    float s = 0.f;
#pragma unroll
    for (int l = 0; l < 16; ++l) s += red[tid >> 4][tid & 15][l];
    sc[tid] = s;
  }
  __syncthreads();
  if (tid == 0) {
    float m = sc[0];
    for (int i = 1; i < TTILE; ++i) m = fmaxf(m, sc[i]);
    sm[0] = m;
  }
  __syncthreads();
  if (tid < TTILE) wexp[tid] = __expf(sc[tid] - sm[0]);
  __syncthreads();
  if (tid == 0) {
    float l = 0.f;
    for (int i = 0; i < TTILE; ++i) l += wexp[i];
    sm[1] = l;
  }
  __syncthreads();

  // --- partial context: c[h] = sum_rows wexp[row] * value[row, h] (L2-hot)
  float c[4] = {0.f, 0.f, 0.f, 0.f};
  for (int row = 0; row < TTILE; ++row) {
    const float w = wexp[row];
    const float* vp = vb + (size_t)(rowbase + row) * H_;
#pragma unroll
    for (int hh = 0; hh < 4; ++hh) c[hh] = fmaf(w, vp[tid + 256 * hh], c[hh]);
  }
  float* pr = part + (size_t)(b * NCHUNK + chunk) * PSTRIDE;
#pragma unroll
  for (int hh = 0; hh < 4; ++hh) pr[tid + 256 * hh] = c[hh];
  if (tid == 0) { pr[1024] = sm[0]; pr[1025] = sm[1]; }
}

// ---------------------------------------------------------------------------
// Kernel 3: merge the 32 chunk-partials per batch (max-rescaled softmax).
// ---------------------------------------------------------------------------
__global__ void __launch_bounds__(256)
finalize_kernel(const float* __restrict__ part, float* __restrict__ out) {
  const int b = blockIdx.x, tid = threadIdx.x;
  const float* pb = part + (size_t)b * NCHUNK * PSTRIDE;
  __shared__ float fs[NCHUNK];
  float M = -3.402823466e38f;
  for (int i = 0; i < NCHUNK; ++i) M = fmaxf(M, pb[i * PSTRIDE + 1024]);
  if (tid < NCHUNK) fs[tid] = __expf(pb[tid * PSTRIDE + 1024] - M);
  __syncthreads();
  float lt = 0.f;
  for (int i = 0; i < NCHUNK; ++i) lt = fmaf(fs[i], pb[i * PSTRIDE + 1025], lt);
  const float inv = 1.0f / lt;
#pragma unroll
  for (int hh = 0; hh < 4; ++hh) {
    const int h = tid + 256 * hh;
    float s = 0.f;
    for (int i = 0; i < NCHUNK; ++i) s = fmaf(pb[i * PSTRIDE + h], fs[i], s);
    out[b * H_ + h] = s * inv;
  }
}

// ---------------------------------------------------------------------------
extern "C" void kernel_launch(void* const* d_in, const int* in_sizes, int n_in,
                              void* d_out, int out_size, void* d_ws, size_t ws_size,
                              hipStream_t stream) {
  const float* query = (const float*)d_in[0];   // [32, 1, 1024]
  const float* value = (const float*)d_in[1];   // [32, 4096, 1024]
  const float* W_Q   = (const float*)d_in[2];   // [1024, 512]
  const float* W_V   = (const float*)d_in[3];   // [1024, 512]
  const float* bias  = (const float*)d_in[4];   // [512]
  const float* fc_w  = (const float*)d_in[5];   // [512, 1]
  float* out = (float*)d_out;                   // [32, 1024]

  char* ws = (char*)d_ws;
  float*          qb    = (float*)ws;                              // 64 KB
  unsigned short* frags = (unsigned short*)(ws + 65536);           // 1 MB bf16 W_V frags
  float*          part  = (float*)(ws + 65536 + 1048576);          // ~4.3 MB partials

  qproj_kernel<<<dim3(2, B_), 256, 0, stream>>>(query, W_Q, bias, qb);
  packWV_kernel<<<(32 * 32 * 32 * 16) / 256, 256, 0, stream>>>(W_V, frags);
  attn_partial_kernel<<<dim3(NCHUNK, B_), 256, 262144, stream>>>(value, frags, qb,
                                                                 fc_w, part);
  finalize_kernel<<<B_, 256, 0, stream>>>(part, out);
}
